// PSAT_25898652795474
// MI455X (gfx1250) — compile-verified, matrix operand order checked by reading, not run on hardware
//
#include <hip/hip_runtime.h>
#include <cstdint>
#include <cstddef>

// ---------------- CDNA5 WMMA types ----------------
typedef __bf16 v16bf __attribute__((ext_vector_type(16)));
typedef float  v8f   __attribute__((ext_vector_type(8)));
typedef unsigned int v4u  __attribute__((ext_vector_type(4)));
typedef int          v8i32 __attribute__((ext_vector_type(8)));
typedef int          v4i32 __attribute__((ext_vector_type(4)));

union FragBF { v16bf v; unsigned int u[8]; };

#ifndef USE_TDM
#if __has_builtin(__builtin_amdgcn_tensor_load_to_lds)
#define USE_TDM 1
#else
#define USE_TDM 0
#endif
#endif

__device__ __forceinline__ unsigned short f2bf(float f) {
  unsigned int u = __float_as_uint(f);
  unsigned int r = (u + 0x7FFFu + ((u >> 16) & 1u)) >> 16;   // RNE
  return (unsigned short)r;
}

// Fragment gather from LDS for 16-bit WMMA operands.
// A tile stored [M][K] (stride elems), B tile stored transposed [N][K].
// Per ISA 7.12.2: lane lm = row, element v -> K = (v>>2)*16 + half*8 + (v&3)*2.
__device__ __forceinline__ void load_frag(FragBF& f, const unsigned short* rowbase,
                                          int stride, int lane) {
  const int lm = lane & 15, half = lane >> 4;
  const unsigned short* p = rowbase + lm * stride + (half << 3);
#pragma unroll
  for (int v = 0; v < 8; ++v) {
    int k = ((v >> 2) << 4) + ((v & 3) << 1);
    f.u[v] = *(const unsigned int*)(p + k);
  }
}

#if USE_TDM
// TDM: DMA a 128x32 bf16 tile (rows x K) from a row-major [.,K] tensor into
// LDS, padding each 32-element (16-dword) row with 4 dwords so the LDS row
// stride is 40 ushorts (= AST below).  D# per CDNA5 ISA ch.8.
__device__ __forceinline__ void tdm_load_tile_a(const unsigned short* gsrc,
                                                unsigned int lds_off, int K) {
  unsigned long long ga = (unsigned long long)(uintptr_t)gsrc;
  v4u g0;
  g0.x = 1u;                                            // count=1 (valid), user mode
  g0.y = lds_off;                                       // lds_addr (bytes)
  g0.z = (unsigned)(ga & 0xFFFFFFFFu);                  // global_addr[31:0]
  g0.w = (unsigned)((ga >> 32) & 0x01FFFFFFu) | (2u << 30);  // addr[56:32] | type=2
  v8i32 g1;
  // data_size=1(2B) | pad_enable | pad_interval=3(16dw) | pad_amount=3(4dw)
  g1[0] = (int)((1u << 16) | (1u << 20) | (3u << 22) | (3u << 25));
  g1[1] = (int)(((unsigned)K & 0xFFFFu) << 16);         // tensor_dim0[15:0]
  g1[2] = (int)(((unsigned)K >> 16) | (128u << 16));    // tensor_dim0[31:16] | tensor_dim1[15:0]=128
  g1[3] = (int)(32u << 16);                             // tensor_dim1[31:16]=0 | tile_dim0=32
  g1[4] = 128;                                          // tile_dim1=128, tile_dim2=0
  g1[5] = K;                                            // tensor_dim0_stride[31:0]
  g1[6] = 0;                                            // stride[47:32] | dim1_stride lo
  g1[7] = 0;
  v4i32 z4 = {0, 0, 0, 0};
#if defined(__clang_major__) && __clang_major__ >= 23
  v8i32 z8 = {0, 0, 0, 0, 0, 0, 0, 0};
  __builtin_amdgcn_tensor_load_to_lds(g0, g1, z4, z4, z8, 0);
#else
  __builtin_amdgcn_tensor_load_to_lds(g0, g1, z4, z4, 0);
#endif
}
#endif

// ---------------- elementwise helpers ----------------
__global__ void k_f32_to_bf16(const float* __restrict__ x,
                              unsigned short* __restrict__ y, int n) {
  int i = blockIdx.x * 256 + threadIdx.x;
  if (i < n) y[i] = f2bf(x[i]);
}

__global__ void k_init_slots(const float* __restrict__ noise,
                             const float* __restrict__ mu,
                             const float* __restrict__ log_sigma,
                             float* __restrict__ slots, int n) {
  int i = blockIdx.x * 256 + threadIdx.x;
  if (i < n) {
    int e = i & 511;
    slots[i] = mu[e] + __expf(log_sigma[e]) * noise[i];
  }
}

// LayerNorm over D=512; optional addend (residual); optional fp32/bf16 outs.
__global__ __launch_bounds__(256) void k_layernorm(
    const float* __restrict__ x, const float* __restrict__ add,
    const float* __restrict__ g, const float* __restrict__ bb,
    float* __restrict__ outf, unsigned short* __restrict__ outb) {
  __shared__ float red[256];
  __shared__ float s_mean, s_rstd;
  const int row = blockIdx.x, t = threadIdx.x;
  const size_t base = (size_t)row * 512;
  float v0 = x[base + t], v1 = x[base + t + 256];
  if (add) { v0 += add[base + t]; v1 += add[base + t + 256]; }
  red[t] = v0 + v1;
  __syncthreads();
  for (int off = 128; off; off >>= 1) {
    if (t < off) red[t] += red[t + off];
    __syncthreads();
  }
  if (t == 0) s_mean = red[0] * (1.0f / 512.0f);
  __syncthreads();
  const float mean = s_mean;
  const float d0 = v0 - mean, d1 = v1 - mean;
  red[t] = d0 * d0 + d1 * d1;
  __syncthreads();
  for (int off = 128; off; off >>= 1) {
    if (t < off) red[t] += red[t + off];
    __syncthreads();
  }
  if (t == 0) s_rstd = rsqrtf(red[0] * (1.0f / 512.0f) + 1e-5f);
  __syncthreads();
  const float rstd = s_rstd;
  const float y0 = d0 * rstd * g[t] + bb[t];
  const float y1 = d1 * rstd * g[t + 256] + bb[t + 256];
  if (outf) { outf[base + t] = y0; outf[base + t + 256] = y1; }
  if (outb) { outb[base + t] = f2bf(y0); outb[base + t + 256] = f2bf(y1); }
}

// ---------------- main bf16 WMMA GEMM (software pipelined) ----------------
// C[M,N] = A[M,K] @ B[K,N] (+bias[n]) (+res[m,n]); A,B bf16, C fp32 or bf16.
// Requires M%128==0, N%64==0, K%32==0 (true for all call sites).
// 256 thr = 8 waves (4Mx2N), each wave 32x32 = 2x2 WMMA frags.  Double-
// buffered LDS: tile k+1 is fetched (TDM for A, regs for B) while tile k
// feeds the WMMAs; one barrier per K step.
#define AST 40
#define BST 40
__global__ __launch_bounds__(256) void k_gemm(
    const unsigned short* __restrict__ A, const unsigned short* __restrict__ B,
    float* __restrict__ Cf, unsigned short* __restrict__ Cb,
    const float* __restrict__ bias, const float* __restrict__ res,
    int M, int N, int K) {
  __shared__ __align__(16) unsigned short As[2][128 * AST];
  __shared__ __align__(16) unsigned short Bs[2][64 * BST];
  const int t = threadIdx.x;
  const int wid = t >> 5, lane = t & 31, lm = lane & 15, half = lane >> 4;
  const int wm = wid >> 1, wn = wid & 1;
  const int m0 = blockIdx.y * 128;
  const int n0 = blockIdx.x * 64;
  const int kr = t >> 3, n8 = (t & 7) * 8;   // this thread's B stage slot

  v8f acc00 = {}, acc01 = {}, acc10 = {}, acc11 = {};

  // ---- prologue: stage tile 0 into buffer 0 ----
  uint4 bReg = *(const uint4*)(B + (size_t)kr * N + n0 + n8);
#if USE_TDM
  if (wid == 0)
    tdm_load_tile_a(A + (size_t)m0 * K,
                    (unsigned)(uintptr_t)&As[0][0], K);
#else
  uint4 aReg0, aReg1;
  {
    int r0 = t >> 2, s0 = t & 3;
    aReg0 = *(const uint4*)(A + (size_t)(m0 + r0) * K + s0 * 8);
    int r1 = (t + 256) >> 2, s1 = t & 3;
    aReg1 = *(const uint4*)(A + (size_t)(m0 + r1) * K + s1 * 8);
  }
#endif
  {
    const unsigned short* p = (const unsigned short*)&bReg;
#pragma unroll
    for (int e = 0; e < 8; ++e) Bs[0][(n8 + e) * BST + kr] = p[e];
  }
#if !USE_TDM
  *(uint4*)(&As[0][(t >> 2) * AST + (t & 3) * 8]) = aReg0;
  *(uint4*)(&As[0][((t + 256) >> 2) * AST + (t & 3) * 8]) = aReg1;
#else
  if (wid == 0) __builtin_amdgcn_s_wait_tensorcnt(0);
#endif
  __syncthreads();

  int buf = 0;
  for (int k0 = 0; k0 < K; k0 += 32) {
    const int nbuf = buf ^ 1;
    const bool more = (k0 + 32) < K;

    // issue next-tile global traffic before compute (overlaps WMMAs below)
    if (more) {
      bReg = *(const uint4*)(B + (size_t)(k0 + 32 + kr) * N + n0 + n8);
#if USE_TDM
      if (wid == 0)
        tdm_load_tile_a(A + (size_t)m0 * K + (k0 + 32),
                        (unsigned)(uintptr_t)&As[nbuf][0], K);
#else
      int r0 = t >> 2, s0 = t & 3;
      aReg0 = *(const uint4*)(A + (size_t)(m0 + r0) * K + (k0 + 32) + s0 * 8);
      int r1 = (t + 256) >> 2;
      aReg1 = *(const uint4*)(A + (size_t)(m0 + r1) * K + (k0 + 32) + s0 * 8);
#endif
    }

    // ---- compute current buffer ----
    FragBF a0, a1, b0, b1;
    load_frag(a0, &As[buf][(wm * 32) * AST], AST, lane);
    load_frag(a1, &As[buf][(wm * 32 + 16) * AST], AST, lane);
    load_frag(b0, &Bs[buf][(wn * 32) * BST], BST, lane);
    load_frag(b1, &Bs[buf][(wn * 32 + 16) * BST], BST, lane);
    acc00 = __builtin_amdgcn_wmma_f32_16x16x32_bf16(false, a0.v, false, b0.v,
                                                    (short)0, acc00, false, false);
    acc01 = __builtin_amdgcn_wmma_f32_16x16x32_bf16(false, a0.v, false, b1.v,
                                                    (short)0, acc01, false, false);
    acc10 = __builtin_amdgcn_wmma_f32_16x16x32_bf16(false, a1.v, false, b0.v,
                                                    (short)0, acc10, false, false);
    acc11 = __builtin_amdgcn_wmma_f32_16x16x32_bf16(false, a1.v, false, b1.v,
                                                    (short)0, acc11, false, false);

    // ---- commit staged tile to the other buffer ----
    if (more) {
      const unsigned short* p = (const unsigned short*)&bReg;
#pragma unroll
      for (int e = 0; e < 8; ++e) Bs[nbuf][(n8 + e) * BST + kr] = p[e];
#if !USE_TDM
      *(uint4*)(&As[nbuf][(t >> 2) * AST + (t & 3) * 8]) = aReg0;
      *(uint4*)(&As[nbuf][((t + 256) >> 2) * AST + (t & 3) * 8]) = aReg1;
#endif
    }
#if USE_TDM
    if (wid == 0) __builtin_amdgcn_s_wait_tensorcnt(0);
#endif
    __syncthreads();
    buf = nbuf;
  }

  auto store_tile = [&](v8f& c, int mb, int nb) {
    const int n = n0 + nb + lm;
#pragma unroll
    for (int v = 0; v < 8; ++v) {
      const int m = m0 + mb + half * 8 + v;    // C layout: vgpr v -> M = half*8+v
      float val = c[v];
      if (bias) val += bias[n];
      if (res)  val += res[(size_t)m * N + n];
      if (Cb) Cb[(size_t)m * N + n] = f2bf(val);
      else    Cf[(size_t)m * N + n] = val;
    }
  };
  store_tile(acc00, wm * 32, wn * 32);
  store_tile(acc01, wm * 32, wn * 32 + 16);
  store_tile(acc10, wm * 32 + 16, wn * 32);
  store_tile(acc11, wm * 32 + 16, wn * 32 + 16);
}

// ---------------- batched upd GEMM: upd[b] = attn[b](8x1024) @ v_[b](1024x512)
// M padded 8->16; attn converted to bf16 during LDS staging; v_ already bf16
// (L2-resident).  Grid (512/128, 32).  8 waves, each one 16x16 column tile.
__global__ __launch_bounds__(256) void k_upd_gemm(
    const float* __restrict__ attn, const unsigned short* __restrict__ V,
    float* __restrict__ upd) {
  __shared__ __align__(16) unsigned short As[16 * AST];
  __shared__ __align__(16) unsigned short Bs[128 * BST];
  const int t = threadIdx.x;
  const int b = blockIdx.y;
  const int n0 = blockIdx.x * 128;
  const int wid = t >> 5, lane = t & 31, lm = lane & 15, half = lane >> 4;
  const float* Ab = attn + (size_t)b * 8 * 1024;
  const unsigned short* Vb = V + (size_t)b * 1024 * 512;

  v8f acc = {};
  for (int k0 = 0; k0 < 1024; k0 += 32) {
#pragma unroll
    for (int i = 0; i < 2; ++i) {
      int idx = t + i * 256;               // 0..511 -> 16 rows x 32 cols
      int r = idx >> 5, c = idx & 31;
      As[r * AST + c] = (r < 8) ? f2bf(Ab[r * 1024 + k0 + c]) : (unsigned short)0;
    }
#pragma unroll
    for (int i = 0; i < 2; ++i) {
      int idx = t + i * 256;               // 0..511 -> 32 rows x 16 segs
      int kr = idx >> 4, seg = idx & 15;
      uint4 val = *(const uint4*)(Vb + (size_t)(k0 + kr) * 512 + n0 + seg * 8);
      const unsigned short* p = (const unsigned short*)&val;
#pragma unroll
      for (int e = 0; e < 8; ++e) Bs[(seg * 8 + e) * BST + kr] = p[e];
    }
    __syncthreads();
    FragBF a, bb;
    load_frag(a, As, AST, lane);
    load_frag(bb, Bs + (wid * 16) * BST, BST, lane);
    acc = __builtin_amdgcn_wmma_f32_16x16x32_bf16(false, a.v, false, bb.v,
                                                  (short)0, acc, false, false);
    __syncthreads();
  }
  if (half == 0) {                          // rows 0..7 valid, 8..15 were pad
    const int n = n0 + wid * 16 + lm;
#pragma unroll
    for (int v = 0; v < 8; ++v)
      upd[((size_t)b * 8 + v) * 512 + n] = acc[v];
  }
}

// ---------------- slot-competition attention (tiny) ----------------
__global__ void k_dots_softmax(const float* __restrict__ q,
                               const float* __restrict__ kk,
                               float* __restrict__ attnS) {
  __shared__ float sd[8][8];
  const int b = blockIdx.x, t = threadIdx.x;
  const int i = t >> 3, j = t & 7;
  const float* qr = q + ((size_t)b * 8 + i) * 512;
  const float* kr = kk + ((size_t)b * 8 + j) * 512;
  float s = 0.f;
  for (int e = 0; e < 512; ++e) s += qr[e] * kr[e];
  sd[i][j] = s * 0.044194173824159216f;     // 512^-0.5
  __syncthreads();
  if (t < 8) {                              // softmax over slot axis (column j=t)
    int jj = t;
    float mx = sd[0][jj];
    for (int ii = 1; ii < 8; ++ii) mx = fmaxf(mx, sd[ii][jj]);
    float sum = 0.f;
    for (int ii = 0; ii < 8; ++ii) { float e = __expf(sd[ii][jj] - mx); sd[ii][jj] = e; sum += e; }
    for (int ii = 0; ii < 8; ++ii) sd[ii][jj] = sd[ii][jj] / sum + 1e-8f;
  }
  __syncthreads();
  if (t < 8) {                              // renormalize row i=t over j
    int ii = t;
    float sum = 0.f;
    for (int jj = 0; jj < 8; ++jj) sum += sd[ii][jj];
    for (int jj = 0; jj < 8; ++jj) attnS[b * 64 + ii * 8 + jj] = sd[ii][jj] / sum;
  }
}

// attn[b,i,t] = sum_j attnS[b,i,j] * prev_attn[b,t,j]
__global__ __launch_bounds__(256) void k_attn_compose(
    const float* __restrict__ attnS, const float* __restrict__ prev_attn,
    float* __restrict__ attn) {
  __shared__ float sA[64];
  const int b = blockIdx.x, t = threadIdx.x;
  if (t < 64) sA[t] = attnS[b * 64 + t];
  __syncthreads();
  for (int idx = t; idx < 8192; idx += 256) {
    int i = idx >> 10, col = idx & 1023;
    const float* pa = prev_attn + (size_t)b * 8192 + col * 8;
    float s = 0.f;
#pragma unroll
    for (int j = 0; j < 8; ++j) s += sA[i * 8 + j] * pa[j];
    attn[(size_t)b * 8192 + i * 1024 + col] = s;
  }
}

// ---------------- 8-token / 8-head MHSA core (tiny) ----------------
__global__ void k_mhsa(const float* __restrict__ qa, const float* __restrict__ ka,
                       const float* __restrict__ va, unsigned short* __restrict__ o) {
  __shared__ float qs[8][64], ks[8][64], vs[8][64], att[8][8];
  const int b = blockIdx.x >> 3, h = blockIdx.x & 7, t = threadIdx.x;
  for (int idx = t; idx < 512; idx += 64) {
    int i = idx >> 6, e = idx & 63;
    size_t base = ((size_t)b * 8 + i) * 512 + h * 64 + e;
    qs[i][e] = qa[base]; ks[i][e] = ka[base]; vs[i][e] = va[base];
  }
  __syncthreads();
  {
    int i = t >> 3, j = t & 7;
    float s = 0.f;
#pragma unroll 8
    for (int e = 0; e < 64; ++e) s += qs[i][e] * ks[j][e];
    att[i][j] = s * 0.125f;                 // 64^-0.5
  }
  __syncthreads();
  if (t < 8) {
    int i = t;
    float mx = att[i][0];
    for (int j = 1; j < 8; ++j) mx = fmaxf(mx, att[i][j]);
    float sum = 0.f;
    for (int j = 0; j < 8; ++j) { float e = __expf(att[i][j] - mx); att[i][j] = e; sum += e; }
    float inv = 1.0f / sum;
    for (int j = 0; j < 8; ++j) att[i][j] *= inv;
  }
  __syncthreads();
  for (int idx = t; idx < 512; idx += 64) {
    int i = idx >> 6, e = idx & 63;
    float s = 0.f;
#pragma unroll
    for (int j = 0; j < 8; ++j) s += att[i][j] * vs[j][e];
    o[((size_t)b * 8 + i) * 512 + h * 64 + e] = f2bf(s);
  }
}

// SwiGLU: out[r,c] = u * silu(gate),  u=ff[r,c], gate=ff[r,2048+c]; bf16 out.
__global__ void k_swiglu(const float* __restrict__ ff,
                         unsigned short* __restrict__ out) {
  int idx = blockIdx.x * 256 + threadIdx.x;
  if (idx < 256 * 2048) {
    int r = idx >> 11, c = idx & 2047;
    float u = ff[(size_t)r * 4096 + c];
    float g = ff[(size_t)r * 4096 + 2048 + c];
    out[idx] = f2bf(u * g / (1.0f + __expf(-g)));
  }
}

// [b,8,1024] -> [b,1024,8]
__global__ __launch_bounds__(256) void k_transpose_attn(
    const float* __restrict__ attn, float* __restrict__ out) {
  const int b = blockIdx.x, t = threadIdx.x;
  for (int idx = t; idx < 8192; idx += 256) {
    int col = idx >> 3, i = idx & 7;
    out[(size_t)b * 8192 + col * 8 + i] = attn[(size_t)b * 8192 + i * 1024 + col];
  }
}

// ---------------- host orchestration ----------------
extern "C" void kernel_launch(void* const* d_in, const int* in_sizes, int n_in,
                              void* d_out, int out_size, void* d_ws, size_t ws_size,
                              hipStream_t stream) {
  const float* features  = (const float*)d_in[0];
  const float* prevslots = (const float*)d_in[1];
  const float* prev_attn = (const float*)d_in[2];
  const float* noise     = (const float*)d_in[3];
  const float* init_mu   = (const float*)d_in[4];
  const float* init_ls   = (const float*)d_in[5];
  const float* Wk  = (const float*)d_in[6];
  const float* Wv  = (const float*)d_in[7];
  const float* Wq  = (const float*)d_in[8];
  const float* g_in = (const float*)d_in[9],  *g_ps = (const float*)d_in[10];
  const float* g_s  = (const float*)d_in[11], *g_ica= (const float*)d_in[12];
  const float* g_a  = (const float*)d_in[13], *g_f  = (const float*)d_in[14];
  const float* g_fin= (const float*)d_in[15];
  const float* b_in = (const float*)d_in[16], *b_ps = (const float*)d_in[17];
  const float* b_s  = (const float*)d_in[18], *b_ica= (const float*)d_in[19];
  const float* b_a  = (const float*)d_in[20], *b_f  = (const float*)d_in[21];
  const float* b_fin= (const float*)d_in[22];
  const float* Waq = (const float*)d_in[23];
  const float* Wak = (const float*)d_in[24];
  const float* Wav = (const float*)d_in[25];
  const float* Wao = (const float*)d_in[26];
  const float* bao = (const float*)d_in[27];
  const float* Wff1= (const float*)d_in[28];
  const float* bff1= (const float*)d_in[29];
  const float* Wff2= (const float*)d_in[30];
  const float* bff2= (const float*)d_in[31];

  char* base = (char*)d_ws;
  size_t off = 0;
  auto alloc = [&](size_t bytes) -> void* {
    void* p = base + off;
    off = (off + bytes + 255) & ~(size_t)255;
    return p;
  };
  typedef unsigned short us;
  us* feats_bf = (us*)alloc(32768ull * 512 * 2);
  us* v_bf     = (us*)alloc(32768ull * 512 * 2);
  us* ps_bf    = (us*)alloc(256ull * 512 * 2);
  us* Wk_bf  = (us*)alloc(262144ull * 2);
  us* Wv_bf  = (us*)alloc(262144ull * 2);
  us* Wq_bf  = (us*)alloc(262144ull * 2);
  us* Waq_bf = (us*)alloc(262144ull * 2);
  us* Wak_bf = (us*)alloc(262144ull * 2);
  us* Wav_bf = (us*)alloc(262144ull * 2);
  us* Wao_bf = (us*)alloc(262144ull * 2);
  us* Wff1_bf= (us*)alloc(2097152ull * 2);
  us* Wff2_bf= (us*)alloc(1048576ull * 2);
  float* k_f    = (float*)alloc(131072ull * 4);
  float* slotsA = (float*)alloc(131072ull * 4);
  float* slotsB = (float*)alloc(131072ull * 4);
  us*    act_bf = (us*)alloc(131072ull * 2);     // reused: ln_s / h_a / h_f
  float* q_f    = (float*)alloc(131072ull * 4);
  float* attnS  = (float*)alloc(2048ull * 4);
  float* attn_f = (float*)alloc(262144ull * 4);
  float* upd_f  = (float*)alloc(131072ull * 4);
  float* x_f    = (float*)alloc(131072ull * 4);
  float* x2_f   = (float*)alloc(131072ull * 4);
  float* x3_f   = (float*)alloc(131072ull * 4);
  float* qa_f   = (float*)alloc(131072ull * 4);
  float* ka_f   = (float*)alloc(131072ull * 4);
  float* va_f   = (float*)alloc(131072ull * 4);
  us*    o_bf   = (us*)alloc(131072ull * 2);
  float* ff_f   = (float*)alloc(1048576ull * 4);
  us*    ffm_bf = (us*)alloc(524288ull * 2);
  (void)ws_size; (void)n_in; (void)in_sizes; (void)out_size;

  auto conv = [&](const float* s, us* d, int n) {
    k_f32_to_bf16<<<(n + 255) / 256, 256, 0, stream>>>(s, d, n);
  };
  auto gemm = [&](const us* A, const us* B, float* Cf, us* Cb,
                  const float* bias, const float* res, int M, int N, int K) {
    dim3 g(N / 64, (M + 127) / 128);
    k_gemm<<<g, 256, 0, stream>>>(A, B, Cf, Cb, bias, res, M, N, K);
  };

  // ---- preamble ----
  conv(Wk, Wk_bf, 262144);   conv(Wv, Wv_bf, 262144);   conv(Wq, Wq_bf, 262144);
  conv(Waq, Waq_bf, 262144); conv(Wak, Wak_bf, 262144); conv(Wav, Wav_bf, 262144);
  conv(Wao, Wao_bf, 262144); conv(Wff1, Wff1_bf, 2097152); conv(Wff2, Wff2_bf, 1048576);
  k_layernorm<<<32768, 256, 0, stream>>>(features, nullptr, g_in, b_in, nullptr, feats_bf);
  k_layernorm<<<256, 256, 0, stream>>>(prevslots, nullptr, g_ps, b_ps, nullptr, ps_bf);
  k_init_slots<<<512, 256, 0, stream>>>(noise, init_mu, init_ls, slotsA, 131072);
  gemm(ps_bf, Wk_bf, k_f, nullptr, nullptr, nullptr, 256, 512, 512);
  gemm(feats_bf, Wv_bf, nullptr, v_bf, nullptr, nullptr, 32768, 512, 512);  // v_ bf16, L2-resident

  // ---- 4 slot-attention steps ----
  float* cur = slotsA;
  float* nxt = slotsB;
  for (int it = 0; it < 4; ++it) {
    k_layernorm<<<256, 256, 0, stream>>>(cur, nullptr, g_s, b_s, nullptr, act_bf);
    gemm(act_bf, Wq_bf, q_f, nullptr, nullptr, nullptr, 256, 512, 512);
    k_dots_softmax<<<32, 64, 0, stream>>>(q_f, k_f, attnS);
    k_attn_compose<<<32, 256, 0, stream>>>(attnS, prev_attn, attn_f);
    k_upd_gemm<<<dim3(4, 32), 256, 0, stream>>>(attn_f, v_bf, upd_f);
    k_layernorm<<<256, 256, 0, stream>>>(cur, upd_f, g_ica, b_ica, x_f, nullptr);
    // encoder: pre-norm MHSA
    k_layernorm<<<256, 256, 0, stream>>>(x_f, nullptr, g_a, b_a, nullptr, act_bf);
    gemm(act_bf, Waq_bf, qa_f, nullptr, nullptr, nullptr, 256, 512, 512);
    gemm(act_bf, Wak_bf, ka_f, nullptr, nullptr, nullptr, 256, 512, 512);
    gemm(act_bf, Wav_bf, va_f, nullptr, nullptr, nullptr, 256, 512, 512);
    k_mhsa<<<256, 64, 0, stream>>>(qa_f, ka_f, va_f, o_bf);
    gemm(o_bf, Wao_bf, x2_f, nullptr, bao, x_f, 256, 512, 512);
    // encoder: pre-norm SwiGLU FF
    k_layernorm<<<256, 256, 0, stream>>>(x2_f, nullptr, g_f, b_f, nullptr, act_bf);
    gemm(act_bf, Wff1_bf, ff_f, nullptr, bff1, nullptr, 256, 4096, 512);
    k_swiglu<<<2048, 256, 0, stream>>>(ff_f, ffm_bf);
    gemm(ffm_bf, Wff2_bf, x3_f, nullptr, bff2, x2_f, 256, 512, 2048);
    float* dst = (it < 3) ? nxt : (float*)d_out;       // final slots -> d_out
    k_layernorm<<<256, 256, 0, stream>>>(x3_f, nullptr, g_fin, b_fin, dst, nullptr);
    if (it < 3) { float* tmp = cur; cur = nxt; nxt = tmp; }
  }
  // final attn map [b,8,1024] -> [b,1024,8]
  k_transpose_attn<<<32, 256, 0, stream>>>(attn_f, (float*)d_out + 131072);
}
